// MACrossAttention_25563645345893
// MI455X (gfx1250) — matrix operand order repaired and checked by reading, not compiled, not run on hardware
//
#include <hip/hip_runtime.h>
#include <math.h>

// ---------------------------------------------------------------------------
// MI455X (gfx1250) implementation of MACrossAttention forward.
//  - criss-cross attention with split (two-part) softmax, PV via bf16 WMMA
//  - qkv 1x1 projection as a bf16 WMMA GEMM (pre-swizzled B operand)
//  - WMMA fragments loaded as contiguous 16B LDS vectors (ds_load_b128)
//  - V tile staged with global_load_async_to_lds_b128 (ASYNCcnt)
//  - MWAB: DFT-based spectral branches + conv/GN weight generators
// ---------------------------------------------------------------------------

#define USE_ASYNC_LDS 1   // global_load_async_to_lds_b128 + s_wait_asynccnt

typedef __bf16 bf16;
typedef bf16  v16bf __attribute__((ext_vector_type(16)));
typedef bf16  v8bf  __attribute__((ext_vector_type(8)));
typedef float v8f   __attribute__((ext_vector_type(8)));

union V16U { v16bf v; v8bf h[2]; };

#define HH   128   // H == W
#define CC   64
#define D8C  8
#define PI_F 3.14159265358979323846f

__device__ inline unsigned bfbits(float f) {        // RNE fp32 -> bf16 bits
  unsigned u = __builtin_bit_cast(unsigned, f);
  return (u + 0x7FFFu + ((u >> 16) & 1u)) >> 16;
}
__device__ inline bf16 f2bf(float f) {
  return __builtin_bit_cast(bf16, (unsigned short)bfbits(f));
}
__device__ inline float geluf(float x) { return 0.5f * x * (1.f + erff(x * 0.70710678118654752f)); }

// --------------------------- WMMA fragment loaders -------------------------
// A (16x32 bf16): lane = M; lane's 16 halfwords are the contiguous
// K-run [16*hi, 16*hi+16)  ->  2x ds_load_b128.
__device__ inline v16bf wmma_load_a_bf(const bf16* src, int ld, int lane) {
  const int m = lane & 15, hi = lane >> 4;
  const bf16* row = src + m * ld + 16 * hi;
  V16U u;
  u.h[0] = *(const v8bf*)(row);
  u.h[1] = *(const v8bf*)(row + 8);
  return u.v;
}
// B (32x16) with logical B[k][n] = srcT[n][k] (e.g. P^T): lane n reads the
// contiguous run srcT[n][16*hi .. 16*hi+16)  ->  2x ds_load_b128.
__device__ inline v16bf wmma_load_bT(const bf16* srcT, int ld, int lane) {
  const int n = lane & 15, hi = lane >> 4;
  const bf16* row = srcT + n * ld + 16 * hi;
  V16U u;
  u.h[0] = *(const v8bf*)(row);
  u.h[1] = *(const v8bf*)(row + 8);
  return u.v;
}
// Pre-swizzled B tile: 32 lanes x 16 contiguous halfwords.
__device__ inline v16bf wmma_load_sw(const bf16* tileBase, int lane) {
  const v8bf* p = (const v8bf*)(tileBase + lane * 16);
  V16U u;
  u.h[0] = p[0];
  u.h[1] = p[1];
  return u.v;
}

// --------------------------- qkv projection (WMMA) -------------------------
// One block per (b,h) row: 128 pixels x 80 outputs, K=64 (2 bf16 WMMA steps).
__global__ __launch_bounds__(256)
void k_qkv_proj(const float* __restrict__ x,
                const float* __restrict__ qw, const float* __restrict__ qb,
                const float* __restrict__ kw, const float* __restrict__ kb,
                const float* __restrict__ vw, const float* __restrict__ vb_,
                float* __restrict__ qo, float* __restrict__ ko,
                float* __restrict__ vo, float* __restrict__ vcol) {
  const int bh = blockIdx.x, b = bh >> 7, h = bh & 127;
  __shared__ __align__(16) bf16 Xs[HH][CC + 8];       // 72 elem rows (144B, 16B-mult)
  __shared__ __align__(16) bf16 Wsw[10][32][16];      // swizzled B tiles (kt*5+nt)
  __shared__ float bias[80];
  const int tid = threadIdx.x;
  __builtin_prefetch(x + ((long)b * CC * HH + h) * HH, 0, 0);   // global_prefetch_b8
  // weights: combined [k=cin 64][n=cout 80]; store in WMMA register order:
  // lane = 16*(kk>=16)+nn, slot = kk&15  (kk = k%32 within k-tile)
  for (int e = tid; e < CC * 80; e += 256) {
    int n = e % 80, k = e / 80;
    float wval = (n < 8) ? qw[n * CC + k] : (n < 16) ? kw[(n - 8) * CC + k] : vw[(n - 16) * CC + k];
    int kt = k >> 5, kk = k & 31, nt = n >> 4, nn = n & 15;
    int lane = ((kk >> 4) << 4) | nn, slot = kk & 15;
    Wsw[kt * 5 + nt][lane][slot] = f2bf(wval);
  }
  if (tid < 80) bias[tid] = (tid < 8) ? qb[tid] : (tid < 16) ? kb[tid - 8] : vb_[tid - 16];
  // x tile: 128 pixels x 64 ch, packed 2 bf16 per b32 store
  const long xbase = ((long)b * CC * HH + h) * HH;
  for (int e = tid; e < (CC / 2) * HH; e += 256) {
    int p = e & 127, c0 = (e >> 7) * 2;
    unsigned lo = bfbits(x[xbase + (long)c0 * HH * HH + p]);
    unsigned hi = bfbits(x[xbase + (long)(c0 + 1) * HH * HH + p]);
    *(unsigned*)&Xs[p][c0] = lo | (hi << 16);
  }
  __syncthreads();
  const int wv = tid >> 5, lane = tid & 31;
  for (int nt = 0; nt < 5; ++nt) {                    // wave wv owns M-tile wv
    const int mt = wv, nbase = nt * 16;
    v8f acc;
    float bv = bias[nbase + (lane & 15)];
#pragma unroll
    for (int r = 0; r < 8; ++r) acc[r] = bv;
#pragma unroll
    for (int kt = 0; kt < 2; ++kt) {
      v16bf a  = wmma_load_a_bf(&Xs[mt * 16][kt * 32], CC + 8, lane);
      v16bf bm = wmma_load_sw(&Wsw[kt * 5 + nt][0][0], lane);
      acc = __builtin_amdgcn_wmma_f32_16x16x32_bf16(false, a, false, bm, (short)0, acc, false, false);
    }
    const int n = nbase + (lane & 15);
    const int mo = (lane >> 4) * 8;
#pragma unroll
    for (int r = 0; r < 8; ++r) {
      int p = mt * 16 + mo + r;
      float val = acc[r];
      if (n < 8)       qo[(((long)b * D8C + n) * HH + h) * HH + p] = val;
      else if (n < 16) ko[(((long)b * D8C + (n - 8)) * HH + h) * HH + p] = val;
      else {
        int c = n - 16;
        vo  [(((long)b * CC + c) * HH + h) * HH + p] = val;
        vcol[(((long)b * HH + p) * CC + c) * HH + h] = val;   // column-major copy for H-attn
      }
    }
  }
}

// --------------------------- criss-cross attention -------------------------
// MODE 0: H-direction (per (b,x) column, diag masked).  MODE 1: W-direction.
// Produces unnormalized O = exp(E-m)@V plus per-row (m,s) for split softmax.
template<int MODE>
__global__ __launch_bounds__(256)
void k_ccattn(const float* __restrict__ qg, const float* __restrict__ kg,
              const float* __restrict__ vg, float* __restrict__ Og,
              float* __restrict__ mg, float* __restrict__ sg) {
  const int b = blockIdx.x >> 7, t0 = blockIdx.x & 127;
  const long q_sc = (long)HH * HH;
  long qb, q_st, vb2, v_sc, ob, o_sc, o_st, mb, m_st;
  if (MODE == 0) {
    qb = (long)b * D8C * HH * HH + t0;   q_st = HH;
    vb2 = ((long)b * HH + t0) * CC * HH; v_sc = HH;            // vcol, unit j-stride
    ob = (long)b * CC * HH * HH + t0;    o_sc = (long)HH * HH; o_st = HH;
    mb = (long)b * HH * HH + t0;         m_st = HH;
  } else {
    qb = ((long)b * D8C * HH + t0) * HH; q_st = 1;
    vb2 = ((long)b * CC * HH + t0) * HH; v_sc = (long)HH * HH; // standard v, unit j-stride
    ob = ((long)b * CC * HH + t0) * HH;  o_sc = (long)HH * HH; o_st = 1;
    mb = ((long)b * HH + t0) * HH;       m_st = 1;
  }
  __shared__ float qs[D8C][HH];
  __shared__ float ks[D8C][HH];
  __shared__ __align__(16) float vstage[CC * HH];
  __shared__ __align__(16) bf16 Vb[CC][HH + 8];   // 136 elem rows (272B, 16B-mult)
  __shared__ __align__(16) bf16 Pb[HH][HH + 8];
  __shared__ float ms[HH], ss[HH];
  const int tid = threadIdx.x;

  for (int e = tid; e < D8C * HH; e += 256) {
    int c = e >> 7, t = e & 127;
    qs[c][t] = qg[qb + c * q_sc + (long)t * q_st];
    ks[c][t] = kg[qb + c * q_sc + (long)t * q_st];
  }
#if USE_ASYNC_LDS
  // CDNA5 async-copy path: each lane DMA's 16B of V into LDS (ASYNCcnt).
  for (int e = tid; e < (CC * HH) / 4; e += 256) {
    int flat = e * 4;
    int c = flat >> 7, j = flat & 127;
    const float* gp = vg + vb2 + (long)c * v_sc + j;
    unsigned ldsoff = (unsigned)(size_t)&vstage[flat];
    asm volatile("global_load_async_to_lds_b128 %0, %1, off" :: "v"(ldsoff), "v"(gp) : "memory");
  }
  asm volatile("s_wait_asynccnt 0x0" ::: "memory");
  __builtin_amdgcn_s_wait_tensorcnt(0);
#else
  for (int e = tid; e < CC * HH; e += 256) {
    int c = e >> 7, j = e & 127;
    vstage[e] = vg[vb2 + (long)c * v_sc + j];
  }
#endif
  __syncthreads();
  // fp32 -> bf16, packed 2-per-b32 store
  for (int e = tid; e < (CC * HH) / 2; e += 256) {
    int c = e >> 6, j0 = (e & 63) * 2;
    unsigned lo = bfbits(vstage[c * HH + j0]);
    unsigned hi = bfbits(vstage[c * HH + j0 + 1]);
    *(unsigned*)&Vb[c][j0] = lo | (hi << 16);
  }

  // per-row stats (K=8 logits recomputed on the fly; trivial FLOPs)
  if (tid < HH) {
    const int i = tid;
    float m = -3.4e38f;
    for (int j = 0; j < HH; ++j) {
      float e2 = 0.f;
#pragma unroll
      for (int c = 0; c < D8C; ++c) e2 = fmaf(qs[c][i], ks[c][j], e2);
      if (MODE == 0 && j == i) e2 = -1e30f;
      m = fmaxf(m, e2);
    }
    float s = 0.f;
    for (int j = 0; j < HH; ++j) {
      float e2 = 0.f;
#pragma unroll
      for (int c = 0; c < D8C; ++c) e2 = fmaf(qs[c][i], ks[c][j], e2);
      if (MODE == 0 && j == i) e2 = -1e30f;
      s += __expf(e2 - m);
    }
    ms[i] = m; ss[i] = s;
  }
  __syncthreads();
  // P = exp(E - m), bf16, packed 2-per-b32 store
  for (int e = tid; e < (HH * HH) / 2; e += 256) {
    int i = e >> 6, j0 = (e & 63) * 2;
    float mi = ms[i];
    unsigned pk = 0;
#pragma unroll
    for (int t = 0; t < 2; ++t) {
      int j = j0 + t;
      float e2 = 0.f;
#pragma unroll
      for (int c = 0; c < D8C; ++c) e2 = fmaf(qs[c][i], ks[c][j], e2);
      if (MODE == 0 && j == i) e2 = -1e30f;
      pk |= bfbits(__expf(e2 - mi)) << (16 * t);
    }
    *(unsigned*)&Pb[i][j0] = pk;
  }
  __syncthreads();

  // O[c,i] = sum_j V[c,j] P[i,j]   (C=64 x H=128, K=128): 32 tiles, 8 waves.
  const int wv = tid >> 5, lane = tid & 31;
  for (int tt = wv * 4; tt < wv * 4 + 4; ++tt) {
    const int mt = tt >> 3, nt = tt & 7;
    v8f acc = {};
#pragma unroll
    for (int kt = 0; kt < 4; ++kt) {
      v16bf a  = wmma_load_a_bf(&Vb[mt * 16][kt * 32], HH + 8, lane);
      v16bf bm = wmma_load_bT(&Pb[nt * 16][kt * 32], HH + 8, lane);
      acc = __builtin_amdgcn_wmma_f32_16x16x32_bf16(false, a, false, bm, (short)0, acc, false, false);
    }
    const int n = nt * 16 + (lane & 15);
    const int mo = (lane >> 4) * 8;
#pragma unroll
    for (int r = 0; r < 8; ++r) {
      int m2 = mt * 16 + mo + r;
      Og[ob + (long)m2 * o_sc + (long)n * o_st] = acc[r];
    }
  }
  if (tid < HH) {
    mg[mb + (long)tid * m_st] = ms[tid];
    sg[mb + (long)tid * m_st] = ss[tid];
  }
}

// -------- combine split softmax + residual:  yb = gamma*(outH+outW)+x ------
__global__ __launch_bounds__(256)
void k_combine(const float* __restrict__ OH, const float* __restrict__ OW,
               const float* __restrict__ mH, const float* __restrict__ sH,
               const float* __restrict__ mW, const float* __restrict__ sW,
               const float* __restrict__ x, const float* __restrict__ gamma,
               float* __restrict__ yb) {
  long pix = blockIdx.x * 256L + threadIdx.x;
  if (pix >= 8L * HH * HH) return;
  int b = (int)(pix >> 14);
  long r = pix & 16383;
  float mh = mH[pix], mw = mW[pix];
  float m = fmaxf(mh, mw);
  float eh = __expf(mh - m), ew = __expf(mw - m);
  float inv = 1.f / (eh * sH[pix] + ew * sW[pix]);
  float g = gamma[0];
  long base = (long)b * CC * HH * HH + r;
  for (int c = 0; c < CC; ++c) {
    long a = base + (long)c * HH * HH;
    yb[a] = g * (eh * OH[a] + ew * OW[a]) * inv + x[a];
  }
}

// --------------------------- generic small kernels -------------------------
__global__ void k_conv1x1(const float* __restrict__ in, const float* __restrict__ w,
                          float* __restrict__ out, int N, int Cin, int Cout, long S) {
  long idx = blockIdx.x * 256L + threadIdx.x;
  if (idx >= (long)N * Cout * S) return;
  long s = idx % S; int co = (int)((idx / S) % Cout); int n = (int)(idx / (S * Cout));
  const float* ip = in + (long)n * Cin * S + s;
  const float* wp = w + (long)co * Cin;
  float acc = 0.f;
  for (int ci = 0; ci < Cin; ++ci) acc = fmaf(wp[ci], ip[(long)ci * S], acc);
  out[idx] = acc;
}

__global__ __launch_bounds__(256)
void k_gn_stats(const float* __restrict__ in, float* __restrict__ stats, int C, long S, int G) {
  const int n = blockIdx.x / G, g = blockIdx.x % G;
  const int cpg = C / G;
  const float* base = in + ((long)n * C + (long)g * cpg) * S;
  const long cnt = (long)cpg * S;
  float sum = 0.f, sq = 0.f;
  for (long i = threadIdx.x; i < cnt; i += 256) { float v = base[i]; sum += v; sq = fmaf(v, v, sq); }
  __shared__ float rs[256], rq[256];
  rs[threadIdx.x] = sum; rq[threadIdx.x] = sq;
  __syncthreads();
  for (int off = 128; off > 0; off >>= 1) {
    if (threadIdx.x < off) { rs[threadIdx.x] += rs[threadIdx.x + off]; rq[threadIdx.x] += rq[threadIdx.x + off]; }
    __syncthreads();
  }
  if (threadIdx.x == 0) {
    float mean = rs[0] / (float)cnt;
    float var  = rq[0] / (float)cnt - mean * mean;
    stats[2 * blockIdx.x]     = mean;
    stats[2 * blockIdx.x + 1] = rsqrtf(var + 1e-5f);
  }
}

__global__ void k_gn_apply(const float* __restrict__ in, const float* __restrict__ stats,
                           const float* __restrict__ gw, const float* __restrict__ gb,
                           const float* __restrict__ res, float* __restrict__ out,
                           int N, int C, long S, int G, int do_gelu) {
  long idx = blockIdx.x * 256L + threadIdx.x;
  if (idx >= (long)N * C * S) return;
  int c = (int)((idx / S) % C); int n = (int)(idx / (S * C));
  int g = c / (C / G);
  float mean = stats[2 * (n * G + g)], rstd = stats[2 * (n * G + g) + 1];
  float v = (in[idx] - mean) * rstd * gw[c] + gb[c];
  if (do_gelu) v = geluf(v);
  if (res) v += res[idx];
  out[idx] = v;
}

__global__ void k_dw3x3(const float* __restrict__ in, const float* __restrict__ w,
                        float* __restrict__ out, int N, int C, int Hd, int Wd) {
  long idx = blockIdx.x * 256L + threadIdx.x;
  if (idx >= (long)N * C * Hd * Wd) return;
  int xw = (int)(idx % Wd); int yh = (int)((idx / Wd) % Hd); int c = (int)((idx / ((long)Wd * Hd)) % C);
  const float* ip = in + (idx - xw - (long)yh * Wd);
  const float* wp = w + c * 9;
  float acc = 0.f;
  for (int dy = -1; dy <= 1; ++dy) {
    int yy = yh + dy; if (yy < 0 || yy >= Hd) continue;
    for (int dx = -1; dx <= 1; ++dx) {
      int xx = xw + dx; if (xx < 0 || xx >= Wd) continue;
      acc = fmaf(wp[(dy + 1) * 3 + dx + 1], ip[(long)yy * Wd + xx], acc);
    }
  }
  out[idx] = acc;
}

__global__ void k_dw3(const float* __restrict__ in, const float* __restrict__ w,
                      float* __restrict__ out, int N, int C, long L) {
  long idx = blockIdx.x * 256L + threadIdx.x;
  if (idx >= (long)N * C * L) return;
  int xp = (int)(idx % L); int c = (int)((idx / L) % C);
  const float* ip = in + (idx - xp);
  const float* wp = w + c * 3;
  float acc = 0.f;
  for (int d = -1; d <= 1; ++d) {
    int xx = xp + d; if (xx < 0 || xx >= (int)L) continue;
    acc = fmaf(wp[d + 1], ip[xx], acc);
  }
  out[idx] = acc;
}

__global__ void k_slice_get(const float* __restrict__ full, float* __restrict__ dst,
                            int N, int coff, int Cs, long S) {
  long idx = blockIdx.x * 256L + threadIdx.x;
  if (idx >= (long)N * Cs * S) return;
  long s = idx % S; int c = (int)((idx / S) % Cs); int n = (int)(idx / (S * Cs));
  dst[idx] = full[((long)n * CC + coff + c) * S + s];
}
__global__ void k_slice_put(float* __restrict__ full, const float* __restrict__ src,
                            int N, int coff, int Cs, long S) {
  long idx = blockIdx.x * 256L + threadIdx.x;
  if (idx >= (long)N * Cs * S) return;
  long s = idx % S; int c = (int)((idx / S) % Cs); int n = (int)(idx / (S * Cs));
  full[((long)n * CC + coff + c) * S + s] = src[idx];
}

__global__ void k_bilinear(const float* __restrict__ in, float* __restrict__ out,
                           int CHn, int Hi, int Wi, int Ho, int Wo) {
  long idx = blockIdx.x * 256L + threadIdx.x;
  if (idx >= (long)CHn * Ho * Wo) return;
  int ox = (int)(idx % Wo); int oy = (int)((idx / Wo) % Ho); int ch = (int)(idx / ((long)Wo * Ho));
  float fy = (Ho > 1) ? (float)oy * (Hi - 1) / (float)(Ho - 1) : 0.f;
  float fx = (Wo > 1) ? (float)ox * (Wi - 1) / (float)(Wo - 1) : 0.f;
  int y0 = (int)floorf(fy); y0 = min(max(y0, 0), Hi - 1); int y1 = min(y0 + 1, Hi - 1); float ty = fy - (float)y0;
  int x0 = (int)floorf(fx); x0 = min(max(x0, 0), Wi - 1); int x1 = min(x0 + 1, Wi - 1); float tx = fx - (float)x0;
  const float* p = in + (long)ch * Hi * Wi;
  float a = p[y0 * Wi + x0], b2 = p[y0 * Wi + x1], c2 = p[y1 * Wi + x0], d = p[y1 * Wi + x1];
  out[idx] = (a * (1 - tx) + b2 * tx) * (1 - ty) + (c2 * (1 - tx) + d * tx) * ty;
}

// --------------------------- spectral (DFT) kernels ------------------------
// Row addressing for the three MWAB branches inside yb (B,64,128,128):
//  mode 0 (y1): rows over W for (b,a,c) triples, channels  0..15
//  mode 1 (y2): rows over H for (b,w,c) triples, channels 16..31
//  mode 2 (y3): rows over W for (b,c,h) triples, channels 32..47
__device__ inline void row_addr(int mode, int r, long& base, long& st) {
  if (mode == 0)      { int c = r & 15;  int a = (r >> 4) & 127; int b_ = r >> 11;
    base = (((long)b_ * 64 + c) * 128 + a) * 128; st = 1; }
  else if (mode == 1) { int c = r & 15;  int w = (r >> 4) & 127; int b_ = r >> 11;
    base = ((long)b_ * 64 + 16 + c) * 128L * 128 + w; st = 128; }
  else                { int h = r & 127; int c = (r >> 7) & 15;  int b_ = r >> 11;
    base = (((long)b_ * 64 + 32 + c) * 128 + h) * 128; st = 1; }
}

__global__ __launch_bounds__(128)
void k_rdft_rows(const float* __restrict__ in, float* __restrict__ re, float* __restrict__ im, int mode) {
  const int r = blockIdx.x;
  long base, st; row_addr(mode, r, base, st);
  __shared__ float row[128];
  const int tid = threadIdx.x;
  row[tid] = in[base + (long)tid * st];
  __syncthreads();
  if (tid < 65) {
    float sr = 0.f, si = 0.f;
    for (int w2 = 0; w2 < 128; ++w2) {
      float ang = -2.f * PI_F * (float)(tid * w2) / 128.f;
      sr = fmaf(row[w2], cosf(ang), sr);
      si = fmaf(row[w2], sinf(ang), si);
    }
    const float s = 0.08838834764831845f;   // 1/sqrt(128), ortho
    re[r * 65 + tid] = sr * s;
    im[r * 65 + tid] = si * s;
  }
}

__global__ void k_cdft(const float* __restrict__ re_i, const float* __restrict__ im_i,
                       float* __restrict__ re_o, float* __restrict__ im_o,
                       int N2, long nset, float sign) {
  long idx = blockIdx.x * 256L + threadIdx.x;
  if (idx >= nset * N2) return;
  int j = (int)(idx % N2); long s = idx / N2;
  long outer = s / 65; int k = (int)(s % 65);
  long base = outer * (long)N2 * 65 + k;
  float sr = 0.f, si = 0.f;
  for (int t = 0; t < N2; ++t) {
    float ar = re_i[base + (long)t * 65], ai = im_i[base + (long)t * 65];
    float ang = sign * 2.f * PI_F * (float)(j * t) / (float)N2;
    float cs = cosf(ang), sn = sinf(ang);
    sr += ar * cs - ai * sn;
    si += ar * sn + ai * cs;
  }
  float sc = rsqrtf((float)N2);              // ortho
  re_o[base + (long)j * 65] = sr * sc;
  im_o[base + (long)j * 65] = si * sc;
}

__global__ void k_cmul(float* __restrict__ re, float* __restrict__ im,
                       const float* __restrict__ wv, long wImOff, int mode) {
  long idx = blockIdx.x * 256L + threadIdx.x;
  if (idx >= 16384L * 65) return;
  int k = (int)(idx % 65); long r = idx / 65;
  long widx;
  if (mode == 2) { int h = (int)(r & 127); int c = (int)((r >> 7) & 15); widx = ((long)c * 128 + h) * 65 + k; }
  else           { int c = (int)(r & 15); widx = (long)c * 65 + k; }
  float wr = wv[widx], wi = wv[wImOff + widx];
  float ar = re[idx], ai = im[idx];
  re[idx] = ar * wr - ai * wi;
  im[idx] = ar * wi + ai * wr;
}

__global__ __launch_bounds__(128)
void k_irdft_rows(const float* __restrict__ re, const float* __restrict__ im,
                  float* __restrict__ out, int mode) {
  const int r = blockIdx.x;
  __shared__ float rr[65], ri[65];
  const int tid = threadIdx.x;
  if (tid < 65) { rr[tid] = re[r * 65 + tid]; ri[tid] = im[r * 65 + tid]; }
  __syncthreads();
  float acc = rr[0] + ((tid & 1) ? -rr[64] : rr[64]);  // Im of k=0,64 dropped (irfft)
  for (int k = 1; k < 64; ++k) {
    float ang = 2.f * PI_F * (float)(k * tid) / 128.f;
    acc += 2.f * (rr[k] * cosf(ang) - ri[k] * sinf(ang));
  }
  long base, st; row_addr(mode, r, base, st);
  out[base + (long)tid * st] = acc * 0.08838834764831845f;
}

// --------------------------- host orchestration ----------------------------
struct IdwP { const float *w1, *g1w, *g1b, *w2, *g2w, *g2b, *w3, *g3w, *g3b; };

static void launch_conv1x1(const float* in, const float* w, float* out,
                           int N, int Cin, int Cout, long S, hipStream_t st) {
  long total = (long)N * Cout * S;
  k_conv1x1<<<dim3((unsigned)((total + 255) / 256)), dim3(256), 0, st>>>(in, w, out, N, Cin, Cout, S);
}
static void launch_gn(const float* in, float* out, const float* gw, const float* gb, float* statbuf,
                      int N, int C, long S, int G, int gelu_f, const float* res, hipStream_t st) {
  k_gn_stats<<<dim3(N * G), dim3(256), 0, st>>>(in, statbuf, C, S, G);
  long total = (long)N * C * S;
  k_gn_apply<<<dim3((unsigned)((total + 255) / 256)), dim3(256), 0, st>>>(in, statbuf, gw, gb, res, out, N, C, S, G, gelu_f);
}
static void run_idw1d(const float* xin, float* out, int N, int cin, int cout, long L,
                      const IdwP& P, bool sc, float* tA, float* tB, float* stat, hipStream_t st) {
  int hid = 2 * cin;
  launch_conv1x1(xin, P.w1, tA, N, cin, hid, L, st);
  launch_gn(tA, tB, P.g1w, P.g1b, stat, N, hid, L, 4, 1, nullptr, st);
  { long t = (long)N * hid * L; k_dw3<<<dim3((unsigned)((t + 255) / 256)), 256, 0, st>>>(tB, P.w2, tA, N, hid, L); }
  launch_gn(tA, tB, P.g2w, P.g2b, stat, N, hid, L, 4, 1, nullptr, st);
  launch_conv1x1(tB, P.w3, tA, N, hid, cout, L, st);
  launch_gn(tA, out, P.g3w, P.g3b, stat, N, cout, L, 4, 0, sc ? xin : nullptr, st);
}
static void run_idw2d(const float* xin, float* out, int N, int cin, int cout, int Hd, int Wd,
                      const IdwP& P, bool sc, float* tA, float* tB, float* stat, hipStream_t st) {
  int hid = 2 * cin; long S = (long)Hd * Wd;
  launch_conv1x1(xin, P.w1, tA, N, cin, hid, S, st);
  launch_gn(tA, tB, P.g1w, P.g1b, stat, N, hid, S, 2, 1, nullptr, st);
  { long t = (long)N * hid * S; k_dw3x3<<<dim3((unsigned)((t + 255) / 256)), 256, 0, st>>>(tB, P.w2, tA, N, hid, Hd, Wd); }
  launch_gn(tA, tB, P.g2w, P.g2b, stat, N, hid, S, 2, 1, nullptr, st);
  launch_conv1x1(tB, P.w3, tA, N, hid, cout, S, st);
  launch_gn(tA, out, P.g3w, P.g3b, stat, N, cout, S, 4, 0, sc ? xin : nullptr, st);
}
static void run_wg1d(const float* xin, float* out, long L, const IdwP* P,
                     float* u0, float* u1, float* tA, float* tB, float* stat, hipStream_t st) {
  run_idw1d(xin, u0, 2, 16, 32, L, P[0], false, tA, tB, stat, st);
  run_idw1d(u0,  u1, 2, 32, 32, L, P[1], true,  tA, tB, stat, st);
  run_idw1d(u1, out, 2, 32, 16, L, P[2], false, tA, tB, stat, st);
}
static void run_wg2d(const float* xin, float* out, int Hd, int Wd, const IdwP* P,
                     float* u0, float* u1, float* tA, float* tB, float* stat, hipStream_t st) {
  run_idw2d(xin, u0, 2, 16, 32, Hd, Wd, P[0], false, tA, tB, stat, st);
  run_idw2d(u0,  u1, 2, 32, 32, Hd, Wd, P[1], true,  tA, tB, stat, st);
  run_idw2d(u1, out, 2, 32, 16, Hd, Wd, P[2], false, tA, tB, stat, st);
}

extern "C" void kernel_launch(void* const* d_in, const int* in_sizes, int n_in,
                              void* d_out, int out_size, void* d_ws, size_t ws_size,
                              hipStream_t stream) {
  (void)in_sizes; (void)n_in; (void)out_size; (void)ws_size;
  const float* const* in = (const float* const*)d_in;
  int pi = 0;
  const float* X     = in[pi++];
  const float* q_w   = in[pi++]; const float* q_b = in[pi++];
  const float* k_w   = in[pi++]; const float* k_b = in[pi++];
  const float* v_w   = in[pi++]; const float* v_b = in[pi++];
  const float* gamma = in[pi++];
  const float* a_weight = in[pi++];
  const float* b_weight = in[pi++];
  const float* c_weight = in[pi++];
  IdwP wg_a[3], wg_b[3], wg_c[3], dwp;
  auto takeIdw = [&]() {
    IdwP p;
    p.w1 = in[pi++]; p.g1w = in[pi++]; p.g1b = in[pi++];
    p.w2 = in[pi++]; p.g2w = in[pi++]; p.g2b = in[pi++];
    p.w3 = in[pi++]; p.g3w = in[pi++]; p.g3b = in[pi++];
    return p;
  };
  for (int i = 0; i < 3; ++i) wg_a[i] = takeIdw();
  for (int i = 0; i < 3; ++i) wg_b[i] = takeIdw();
  for (int i = 0; i < 3; ++i) wg_c[i] = takeIdw();
  dwp = takeIdw();

  // ---- workspace carve (floats), ~260 MB total ----
  size_t off = 0;
  auto alloc = [&](size_t n) { float* p = (float*)d_ws + off; off += n; return p; };
  const long S2 = (long)HH * HH;                   // 16384
  float* q    = alloc(8L * D8C * S2);
  float* k    = alloc(8L * D8C * S2);
  float* v    = alloc(8L * CC * S2);
  float* vcol = alloc(8L * CC * S2);
  float* OH   = alloc(8L * CC * S2);
  float* OW   = alloc(8L * CC * S2);
  float* mH   = alloc(8L * S2); float* sH = alloc(8L * S2);
  float* mW   = alloc(8L * S2); float* sW = alloc(8L * S2);
  float* yb   = alloc(8L * CC * S2);
  const long FR = 16384L * 65;
  float* tre = alloc(FR); float* tim = alloc(FR);
  float* t2re = alloc(FR); float* t2im = alloc(FR);
  float* big0 = alloc(8L * 32 * S2);
  float* big1 = alloc(8L * 32 * S2);
  float* big2 = alloc(8L * 16 * S2);
  float* x4c  = alloc(8L * 16 * S2);
  float* stat = alloc(1024);
  float* u0 = alloc(2L * 64 * 8320); float* u1 = alloc(2L * 64 * 8320);
  float* uA = alloc(2L * 64 * 8320); float* uB = alloc(2L * 64 * 8320);
  float* aw = alloc(2L * 16 * 65);
  float* bw = alloc(2L * 16 * 65);
  float* cw = alloc(2L * 16 * 128 * 65);
  float* bil = alloc(2L * 16 * 128 * 65);
  float* OUT = (float*)d_out;

  // ---- 1) qkv projection (WMMA) ----
  k_qkv_proj<<<dim3(8 * HH), dim3(256), 0, stream>>>(X, q_w, q_b, k_w, k_b, v_w, v_b, q, k, v, vcol);

  // ---- 2) criss-cross attention, split softmax ----
  k_ccattn<0><<<dim3(8 * HH), dim3(256), 0, stream>>>(q, k, vcol, OH, mH, sH);
  k_ccattn<1><<<dim3(8 * HH), dim3(256), 0, stream>>>(q, k, v,    OW, mW, sW);
  k_combine<<<dim3((unsigned)((8L * S2 + 255) / 256)), dim3(256), 0, stream>>>(
      OH, OW, mH, sH, mW, sW, X, gamma, yb);

  // ---- 3) MWAB branch y4: idw2d(dw, shortcut) on channels 48..63 ----
  { long t = 8L * 16 * S2; unsigned nb = (unsigned)((t + 255) / 256);
    k_slice_get<<<dim3(nb), 256, 0, stream>>>(yb, x4c, 8, 48, 16, S2); }
  run_idw2d(x4c, big2, 8, 16, 16, 128, 128, dwp, true, big0, big1, stat, stream);
  { long t = 8L * 16 * S2; unsigned nb = (unsigned)((t + 255) / 256);
    k_slice_put<<<dim3(nb), 256, 0, stream>>>(OUT, big2, 8, 48, 16, S2); }

  // ---- 4) weight generators (tiny networks on params only) ----
  { long t = 2L * 16 * 65; unsigned nb = (unsigned)((t + 255) / 256);
    k_bilinear<<<dim3(nb), 256, 0, stream>>>(a_weight, bil, 2, 16, 16, 16, 65); }
  run_wg1d(bil, aw, 65, wg_a, u0, u1, uA, uB, stat, stream);
  { long t = 2L * 16 * 65; unsigned nb = (unsigned)((t + 255) / 256);
    k_bilinear<<<dim3(nb), 256, 0, stream>>>(b_weight, bil, 2, 16, 16, 16, 65); }
  run_wg1d(bil, bw, 65, wg_b, u0, u1, uA, uB, stat, stream);
  { long t = 32L * 128 * 65; unsigned nb = (unsigned)((t + 255) / 256);
    k_bilinear<<<dim3(nb), 256, 0, stream>>>(c_weight, bil, 32, 16, 16, 128, 65); }
  run_wg2d(bil, cw, 128, 65, wg_c, u0, u1, uA, uB, stat, stream);

  // ---- 5) spectral branches y1 (ch0-15), y2 (ch16-31), y3 (ch32-47) ----
  for (int mode = 0; mode < 3; ++mode) {
    k_rdft_rows<<<dim3(16384), dim3(128), 0, stream>>>(yb, tre, tim, mode);
    int N2 = (mode == 2) ? 128 : 16;
    long nset = (16384L / N2) * 65L;
    long tot = nset * N2;
    unsigned cb = (unsigned)((tot + 255) / 256);
    k_cdft<<<dim3(cb), dim3(256), 0, stream>>>(tre, tim, t2re, t2im, N2, nset, -1.f);
    const float* wvp = (mode == 0) ? aw : (mode == 1) ? bw : cw;
    long wIm = (mode == 2) ? 16L * 128 * 65 : 16L * 65;
    unsigned mb2 = (unsigned)((16384L * 65 + 255) / 256);
    k_cmul<<<dim3(mb2), 256, 0, stream>>>(t2re, t2im, wvp, wIm, mode);
    k_cdft<<<dim3(cb), dim3(256), 0, stream>>>(t2re, t2im, tre, tim, N2, nset, 1.f);
    k_irdft_rows<<<dim3(16384), dim3(128), 0, stream>>>(tre, tim, OUT, mode);
  }
}
